// Encoder_19834158973085
// MI455X (gfx1250) — matrix operand order repaired and checked by reading, not compiled, not run on hardware
//
#include <hip/hip_runtime.h>
#include <hip/hip_bf16.h>

typedef __attribute__((ext_vector_type(16))) __bf16 v16bf;
typedef __attribute__((ext_vector_type(8)))  __bf16 v8bf;
typedef __attribute__((ext_vector_type(8)))  float  v8f;

#define NNODES 100000
#define NEDGES 200000
#define NGRAPH 5000
#define EPS    1e-05f

// ---------------------------------------------------------------------------
// Generic helpers
// ---------------------------------------------------------------------------
__global__ void k_zero_f32(float* p, long n) {
    long t = (long)blockIdx.x * blockDim.x + threadIdx.x;
    if (t < n) p[t] = 0.0f;
}

__global__ void k_degree(const int* src, float* deg, int E) {
    int t = blockIdx.x * blockDim.x + threadIdx.x;
    if (t < E) atomicAdd(&deg[src[t]], 1.0f);
}

__global__ void k_norm(const float* deg, float* norm, int n) {
    int t = blockIdx.x * blockDim.x + threadIdx.x;
    if (t < n) { float d = deg[t]; norm[t] = (d > 0.0f) ? (1.0f / d) : 0.0f; }
}

// Convert f32 (M x K) -> zero-padded bf16 (Mpad x Kpad), row-major.
__global__ void k_cvt_pad(const float* __restrict__ src, __bf16* __restrict__ dst,
                          int M, int K, int Mpad, int Kpad) {
    long t = (long)blockIdx.x * blockDim.x + threadIdx.x;
    if (t >= (long)Mpad * Kpad) return;
    int r = (int)(t / Kpad);
    int k = (int)(t % Kpad);
    dst[t] = (r < M && k < K) ? (__bf16)src[(size_t)r * K + k] : (__bf16)0.0f;
}

// ---------------------------------------------------------------------------
// Weight packing straight into WMMA B-fragment layout.
// Bfrag element index: (((nt * nkc) + kc) * 32 + lane) * 16 + j
//   col = nt*16 + (lane&15),  k = kc*32 + 8*(lane>>4) + (j&7) + 16*(j>>3)   (zero for k >= K)
// ---------------------------------------------------------------------------
// theta weights: logical B[k=i, col=f*OUT+o] = We[(o*IN+i)*8 + f]
__global__ void k_pack_theta_frag(const float* __restrict__ We, __bf16* __restrict__ Bf,
                                  int IN, int OUT, int Kpad) {
    int nkc = Kpad >> 5;
    int total = (8 * OUT / 16) * nkc * 512;
    int t = blockIdx.x * blockDim.x + threadIdx.x;
    if (t >= total) return;
    int j    = t & 15;
    int lane = (t >> 4) & 31;
    int rem  = t >> 9;
    int kc   = rem % nkc;
    int nt   = rem / nkc;
    int col  = nt * 16 + (lane & 15);
    int k    = kc * 32 + 8 * (lane >> 4) + (j & 7) + 16 * (j >> 3);
    float v = 0.0f;
    if (k < IN) {
        int f = col / OUT, o = col % OUT;
        v = We[(size_t)(o * IN + k) * 8 + f];
    }
    Bf[t] = (__bf16)v;
}

// transposed weights: logical B[k, col] = W[col*K + k]   (for x @ W.T)
__global__ void k_pack_T_frag(const float* __restrict__ W, __bf16* __restrict__ Bf,
                              int K, int NC, int Kpad) {
    int nkc = Kpad >> 5;
    int total = (NC / 16) * nkc * 512;
    int t = blockIdx.x * blockDim.x + threadIdx.x;
    if (t >= total) return;
    int j    = t & 15;
    int lane = (t >> 4) & 31;
    int rem  = t >> 9;
    int kc   = rem % nkc;
    int nt   = rem / nkc;
    int col  = nt * 16 + (lane & 15);
    int k    = kc * 32 + 8 * (lane >> 4) + (j & 7) + 16 * (j >> 3);
    float v = (k < K) ? W[(size_t)col * K + k] : 0.0f;
    Bf[t] = (__bf16)v;
}

// ---------------------------------------------------------------------------
// WMMA GEMM: C(M,NC) = A(Mpad,Kpad) * B(Kpad,NC)
// A: padded bf16 row-major. B: fragment-packed bf16. One wave = 32x32 C tile
// (2x2 WMMA tiles, 4 accumulators, 4 v_wmma per k-step). No bounds checks in
// the k-loop; only the epilogue masks rows >= M.
// ---------------------------------------------------------------------------
#define SHUF16(lo, hi) __builtin_shufflevector(lo, hi, 0,1,2,3,4,5,6,7,8,9,10,11,12,13,14,15)

__global__ void k_wmma_gemm(const __bf16* __restrict__ A,
                            const __bf16* __restrict__ Bf,
                            void* __restrict__ C,
                            int Mpad, int Kpad, int NC, int M,
                            const float* __restrict__ bias,
                            int relu, int out_bf16) {
    int wave = threadIdx.x >> 5;
    int lane = threadIdx.x & 31;
    int half = lane >> 4;
    int l16  = lane & 15;
    int m2 = blockIdx.x * 4 + wave;      // 32-row tile id (uniform per wave)
    int n2 = blockIdx.y;                 // 32-col tile id
    int mbase = m2 * 32;
    if (mbase >= Mpad) return;
    int nbase = n2 * 32;
    int nkc = Kpad >> 5;

    const __bf16* a0p = A + (size_t)(mbase + l16) * Kpad + 8 * half;
    const __bf16* a1p = a0p + (size_t)16 * Kpad;
    const __bf16* b0p = Bf + (((size_t)(2 * n2)     * nkc) * 32 + lane) * 16;
    const __bf16* b1p = Bf + (((size_t)(2 * n2 + 1) * nkc) * 32 + lane) * 16;

    v8f acc00 = {}, acc01 = {}, acc10 = {}, acc11 = {};
    for (int kc = 0; kc < nkc; ++kc) {
        v8bf a0lo = *(const v8bf*)(a0p);        // K = kc*32 + 8*half + 0..7
        v8bf a0hi = *(const v8bf*)(a0p + 16);   // K = kc*32 + 16 + 8*half + 0..7
        v8bf a1lo = *(const v8bf*)(a1p);
        v8bf a1hi = *(const v8bf*)(a1p + 16);
        v16bf b0  = *(const v16bf*)(b0p);
        v16bf b1  = *(const v16bf*)(b1p);
        v16bf a0 = SHUF16(a0lo, a0hi);
        v16bf a1 = SHUF16(a1lo, a1hi);
        acc00 = __builtin_amdgcn_wmma_f32_16x16x32_bf16(false, a0, false, b0, (short)0, acc00, false, false);
        acc01 = __builtin_amdgcn_wmma_f32_16x16x32_bf16(false, a0, false, b1, (short)0, acc01, false, false);
        acc10 = __builtin_amdgcn_wmma_f32_16x16x32_bf16(false, a1, false, b0, (short)0, acc10, false, false);
        acc11 = __builtin_amdgcn_wmma_f32_16x16x32_bf16(false, a1, false, b1, (short)0, acc11, false, false);
        a0p += 32; a1p += 32;
        b0p += 512; b1p += 512;
    }

    int col0 = nbase + l16;
    int col1 = nbase + 16 + l16;
    float bv0 = bias ? bias[col0] : 0.0f;
    float bv1 = bias ? bias[col1] : 0.0f;
#pragma unroll
    for (int r = 0; r < 8; ++r) {
        int row0 = mbase + r + 8 * half;
        int row1 = row0 + 16;
        float v00 = acc00[r] + bv0, v01 = acc01[r] + bv1;
        float v10 = acc10[r] + bv0, v11 = acc11[r] + bv1;
        if (relu) {
            v00 = fmaxf(v00, 0.0f); v01 = fmaxf(v01, 0.0f);
            v10 = fmaxf(v10, 0.0f); v11 = fmaxf(v11, 0.0f);
        }
        if (out_bf16) {
            __bf16* Cb = (__bf16*)C;
            if (row0 < M) { Cb[(size_t)row0 * NC + col0] = (__bf16)v00;
                            Cb[(size_t)row0 * NC + col1] = (__bf16)v01; }
            if (row1 < M) { Cb[(size_t)row1 * NC + col0] = (__bf16)v10;
                            Cb[(size_t)row1 * NC + col1] = (__bf16)v11; }
        } else {
            float* Cf = (float*)C;
            if (row0 < M) { Cf[(size_t)row0 * NC + col0] = v00;
                            Cf[(size_t)row0 * NC + col1] = v01; }
            if (row1 < M) { Cf[(size_t)row1 * NC + col0] = v10;
                            Cf[(size_t)row1 * NC + col1] = v11; }
        }
    }
}

// ---------------------------------------------------------------------------
// Edge message + scatter: agg[dst,o] += norm[src] * sum_f ea[e,f]*Y[src,f,o]
// ---------------------------------------------------------------------------
__global__ void k_edge_msg(const int* __restrict__ src, const int* __restrict__ dst,
                           const float* __restrict__ ea, const __bf16* __restrict__ Y,
                           const float* __restrict__ norm, float* __restrict__ agg,
                           int E, int OUT) {
    long t = (long)blockIdx.x * blockDim.x + threadIdx.x;
    if (t >= (long)E * OUT) return;
    int e = (int)(t / OUT);
    int o = (int)(t % OUT);
    int s = src[e];
    float nv = norm[s];
    const __bf16* y = Y + (size_t)s * 8 * OUT + o;
    const float* a  = ea + (size_t)e * 8;
    float m = 0.0f;
#pragma unroll
    for (int f = 0; f < 8; ++f) m += a[f] * (float)y[(size_t)f * OUT];
    atomicAdd(&agg[(size_t)dst[e] * OUT + o], m * nv);
}

__global__ void k_combine(const float* agg, const float* xr, const float* bias,
                          float* h, long n, int C) {
    long t = (long)blockIdx.x * blockDim.x + threadIdx.x;
    if (t < n) h[t] = agg[t] + xr[t] + bias[t % C];
}

// ---------------------------------------------------------------------------
// BatchNorm
// ---------------------------------------------------------------------------
__global__ void k_bn_stats(const float* __restrict__ x, float* sum, float* sq,
                           int N, int C) {
    int g = blockIdx.x * blockDim.x + threadIdx.x;
    int total = gridDim.x * blockDim.x;
    int c = g % C;
    int stride = total / C;
    float s = 0.0f, q = 0.0f;
    for (int r = g / C; r < N; r += stride) {
        float v = x[(size_t)r * C + c];
        s += v; q += v * v;
    }
    atomicAdd(&sum[c], s);
    atomicAdd(&sq[c], q);
}

__global__ void k_bn_apply(float* x, const float* sum, const float* sq,
                           const float* gam, const float* bet, int N, int C) {
    long t = (long)blockIdx.x * blockDim.x + threadIdx.x;
    if (t >= (long)N * C) return;
    int c = (int)(t % C);
    float mu  = sum[c] / (float)N;
    float var = sq[c] / (float)N - mu * mu;
    float v = (x[t] - mu) * __frsqrt_rn(var + EPS) * gam[c] + bet[c];
    x[t] = fmaxf(v, 0.0f);
}

// ---------------------------------------------------------------------------
// Pooling + head
// ---------------------------------------------------------------------------
__global__ void k_pool_accum(const float* __restrict__ h, const int* __restrict__ batch,
                             float* __restrict__ sums, int N, int C) {
    long t = (long)blockIdx.x * blockDim.x + threadIdx.x;
    if (t >= (long)N * C) return;
    int n = (int)(t / C);
    int c = (int)(t % C);
    atomicAdd(&sums[(size_t)batch[n] * C + c], h[t]);
}

__global__ void k_pool_cnt(const int* batch, float* cnt, int N) {
    int t = blockIdx.x * blockDim.x + threadIdx.x;
    if (t < N) atomicAdd(&cnt[batch[t]], 1.0f);
}

__global__ void k_pool_div(const float* sums, const float* cnt, float* pooled,
                           int G, int C) {
    int t = blockIdx.x * blockDim.x + threadIdx.x;
    if (t >= G * C) return;
    pooled[t] = sums[t] / fmaxf(cnt[t / C], 1.0f);
}

__global__ void k_split(const float* fc2, float* out, int G) {
    int t = blockIdx.x * blockDim.x + threadIdx.x;
    if (t >= G * 128) return;
    int g = t / 128, c = t % 128;
    float v = fc2[t];
    if (c < 64) out[(size_t)g * 64 + c] = v;
    else        out[(size_t)G * 64 + (size_t)g * 64 + (c - 64)] = v;
}

// ---------------------------------------------------------------------------
extern "C" void kernel_launch(void* const* d_in, const int* in_sizes, int n_in,
                              void* d_out, int out_size, void* d_ws, size_t ws_size,
                              hipStream_t stream) {
    const float* x     = (const float*)d_in[0];
    const int*   eidx  = (const int*)d_in[1];
    const float* ea    = (const float*)d_in[2];
    const int*   batch = (const int*)d_in[3];
    const float* We1   = (const float*)d_in[4];
    const float* b1    = (const float*)d_in[5];
    const float* Wr1   = (const float*)d_in[6];
    const float* g1    = (const float*)d_in[7];
    const float* be1   = (const float*)d_in[8];
    const float* We2   = (const float*)d_in[9];
    const float* b2    = (const float*)d_in[10];
    const float* Wr2   = (const float*)d_in[11];
    const float* g2    = (const float*)d_in[12];
    const float* be2   = (const float*)d_in[13];
    const float* Wf1   = (const float*)d_in[14];
    const float* bf1   = (const float*)d_in[15];
    const float* Wf2   = (const float*)d_in[16];
    const float* bf2   = (const float*)d_in[17];
    float* out = (float*)d_out;

    const int N = NNODES, E = NEDGES, G = NGRAPH;
    const int Gpad = 5024;                 // G padded to 32-row multiple
    const int* src = eidx;
    const int* dst = eidx + E;

    // ---- workspace layout ----
    char* base = (char*)d_ws;
    size_t off = 0;
    auto alloc = [&](size_t bytes) -> char* {
        off = (off + 255) & ~(size_t)255;
        char* p = base + off;
        off += bytes;
        return p;
    };
    // regionA: phase 1 = Y1(bf16 N*256) + agg1(f32 N*32) + XR1(f32 N*32);
    //          phase 2 = Y2(bf16 N*512) (Y1/agg1/XR1 dead by then)
    char*  regionA = alloc((size_t)N * 512 * 2);
    __bf16* Y1   = (__bf16*)regionA;
    float*  agg1 = (float*)(regionA + (size_t)N * 256 * 2);
    float*  XR1  = (float*)(regionA + (size_t)N * 256 * 2 + (size_t)N * 32 * 4);
    __bf16* Y2   = (__bf16*)regionA;

    float*  h1    = (float*)alloc((size_t)N * 32 * 4);
    float*  agg2  = (float*)alloc((size_t)N * 64 * 4);
    float*  XR2   = (float*)alloc((size_t)N * 64 * 4);
    float*  h2    = (float*)alloc((size_t)N * 64 * 4);
    float*  deg   = (float*)alloc((size_t)N * 4);
    float*  nrm   = (float*)alloc((size_t)N * 4);
    float*  bnS   = (float*)alloc(64 * 4);
    float*  bnQ   = (float*)alloc(64 * 4);
    float*  psum  = (float*)alloc((size_t)G * 64 * 4);
    float*  pcnt  = (float*)alloc((size_t)G * 4);
    float*  pool  = (float*)alloc((size_t)G * 64 * 4);
    float*  fc1   = (float*)alloc((size_t)G * 128 * 4);
    float*  fc2o  = (float*)alloc((size_t)G * 128 * 4);
    __bf16* xpad  = (__bf16*)alloc((size_t)N * 32 * 2);       // x  -> (N,32)  Kpad 16->32
    __bf16* h1b   = (__bf16*)alloc((size_t)N * 32 * 2);       // h1 -> (N,32)
    __bf16* poolb = (__bf16*)alloc((size_t)Gpad * 64 * 2);    // pooled -> (Gpad,64)
    __bf16* fc1b  = (__bf16*)alloc((size_t)Gpad * 128 * 2);   // fc1 -> (Gpad,128)
    // fragment-packed weights: (NC/16) * (Kpad/32) * 512 bf16
    __bf16* B1f  = (__bf16*)alloc((size_t)16 * 1 * 512 * 2);  // NC=256 Kpad=32
    __bf16* B2f  = (__bf16*)alloc((size_t)32 * 1 * 512 * 2);  // NC=512 Kpad=32
    __bf16* BR1f = (__bf16*)alloc((size_t)2  * 1 * 512 * 2);  // NC=32  Kpad=32
    __bf16* BR2f = (__bf16*)alloc((size_t)4  * 1 * 512 * 2);  // NC=64  Kpad=32
    __bf16* BF1f = (__bf16*)alloc((size_t)8  * 2 * 512 * 2);  // NC=128 Kpad=64
    __bf16* BF2f = (__bf16*)alloc((size_t)8  * 4 * 512 * 2);  // NC=128 Kpad=128
    (void)ws_size;

    auto blocks = [](long n, int bs) { return (int)((n + bs - 1) / bs); };
    auto zero = [&](float* p, long n) {
        k_zero_f32<<<blocks(n, 256), 256, 0, stream>>>(p, n);
    };
    auto gemm = [&](const __bf16* A, const __bf16* B, void* C,
                    int Mpad, int Kpad, int NC, int M,
                    const float* bias, int relu, int obf) {
        dim3 grid((Mpad / 32 + 3) / 4, NC / 32);
        k_wmma_gemm<<<grid, 128, 0, stream>>>(A, B, C, Mpad, Kpad, NC, M, bias, relu, obf);
    };
    auto cvt = [&](const float* s, __bf16* d, int M, int K, int Mp, int Kp) {
        k_cvt_pad<<<blocks((long)Mp * Kp, 256), 256, 0, stream>>>(s, d, M, K, Mp, Kp);
    };

    // ---- degrees, weight packing, A-padding ----
    zero(deg, N);
    k_degree<<<blocks(E, 256), 256, 0, stream>>>(src, deg, E);
    k_norm<<<blocks(N, 256), 256, 0, stream>>>(deg, nrm, N);
    k_pack_theta_frag<<<blocks(16 * 512, 256), 256, 0, stream>>>(We1, B1f, 16, 32, 32);
    k_pack_theta_frag<<<blocks(32 * 512, 256), 256, 0, stream>>>(We2, B2f, 32, 64, 32);
    k_pack_T_frag<<<blocks(2 * 512, 256),  256, 0, stream>>>(Wr1, BR1f, 16, 32, 32);
    k_pack_T_frag<<<blocks(4 * 512, 256),  256, 0, stream>>>(Wr2, BR2f, 32, 64, 32);
    k_pack_T_frag<<<blocks(16 * 512, 256), 256, 0, stream>>>(Wf1, BF1f, 64, 128, 64);
    k_pack_T_frag<<<blocks(32 * 512, 256), 256, 0, stream>>>(Wf2, BF2f, 128, 128, 128);
    cvt(x, xpad, N, 16, N, 32);

    // ---- ECC layer 1 (in=16, out=32) ----
    gemm(xpad, B1f, Y1, N, 32, 256, N, nullptr, 0, 1);      // Y1[n,f,o]
    zero(agg1, (long)N * 32);
    k_edge_msg<<<blocks((long)E * 32, 256), 256, 0, stream>>>(src, dst, ea, Y1, nrm, agg1, E, 32);
    gemm(xpad, BR1f, XR1, N, 32, 32, N, nullptr, 0, 0);     // x @ Wr1.T
    k_combine<<<blocks((long)N * 32, 256), 256, 0, stream>>>(agg1, XR1, b1, h1, (long)N * 32, 32);
    zero(bnS, 64); zero(bnQ, 64);
    k_bn_stats<<<512, 256, 0, stream>>>(h1, bnS, bnQ, N, 32);
    k_bn_apply<<<blocks((long)N * 32, 256), 256, 0, stream>>>(h1, bnS, bnQ, g1, be1, N, 32);
    cvt(h1, h1b, N, 32, N, 32);

    // ---- ECC layer 2 (in=32, out=64) ----  (Y2 reuses regionA)
    gemm(h1b, B2f, Y2, N, 32, 512, N, nullptr, 0, 1);
    zero(agg2, (long)N * 64);
    k_edge_msg<<<blocks((long)E * 64, 256), 256, 0, stream>>>(src, dst, ea, Y2, nrm, agg2, E, 64);
    gemm(h1b, BR2f, XR2, N, 32, 64, N, nullptr, 0, 0);
    k_combine<<<blocks((long)N * 64, 256), 256, 0, stream>>>(agg2, XR2, b2, h2, (long)N * 64, 64);
    zero(bnS, 64); zero(bnQ, 64);
    k_bn_stats<<<512, 256, 0, stream>>>(h2, bnS, bnQ, N, 64);
    k_bn_apply<<<blocks((long)N * 64, 256), 256, 0, stream>>>(h2, bnS, bnQ, g2, be2, N, 64);

    // ---- mean pool over graphs ----
    zero(psum, (long)G * 64); zero(pcnt, G);
    k_pool_accum<<<blocks((long)N * 64, 256), 256, 0, stream>>>(h2, batch, psum, N, 64);
    k_pool_cnt<<<blocks(N, 256), 256, 0, stream>>>(batch, pcnt, N);
    k_pool_div<<<blocks((long)G * 64, 256), 256, 0, stream>>>(psum, pcnt, pool, G, 64);

    // ---- MLP head ----
    cvt(pool, poolb, G, 64, Gpad, 64);
    gemm(poolb, BF1f, fc1, Gpad, 64, 128, G, bf1, 1, 0);    // relu(pooled @ Wf1.T + bf1)
    cvt(fc1, fc1b, G, 128, Gpad, 128);
    gemm(fc1b, BF2f, fc2o, Gpad, 128, 128, G, bf2, 0, 0);   // @ Wf2.T + bf2
    k_split<<<blocks((long)G * 128, 256), 256, 0, stream>>>(fc2o, out, G);
    (void)out_size; (void)n_in; (void)in_sizes;
}
